// PointNet2Encoder_26396869001381
// MI455X (gfx1250) — matrix operand order repaired and checked by reading, not compiled, not run on hardware
//
#include <hip/hip_runtime.h>
#include <hip/hip_bf16.h>

// ---------------------------------------------------------------------------
// PointNet++ encoder for MI455X (gfx1250, wave32, WMMA 16x16x32 f16).
// Fragment maps per CDNA5 ISA 7.12.2:
//   A (16x32 f16): lane row M = lane&15; elems 0..7  -> K = kk + (lane>>4)*8 ..+7
//                                        elems 8..15 -> K = kk + 16 + (lane>>4)*8 ..+7
//   B (32x16 f16): lane col N = lane&15; same K map as A (elems 0..7 -> kk+(lane>>4)*8, 8..15 -> kk+16+(lane>>4)*8)
//   C (16x16 f32): VGPR r holds row m0+r (lanes 0-15) / m0+r+8 (lanes 16-31),
//                  col n0 + (lane&15)
// Weights pre-transposed to [Cout][Kpad] f16 -> B fragment = two contiguous
// 16-byte loads. Each wave owns an N-tile column and a compile-time chunk of
// M-tiles, so one B fragment feeds MCH WMMAs with no divergent guards in the
// K loop (keeps EXEC untouched and codegen straight-line).
// Bias vectors are staged to LDS via global_load_async_to_lds_b32 (ASYNCcnt).
// ---------------------------------------------------------------------------

typedef __attribute__((ext_vector_type(16))) _Float16 v16h;
typedef __attribute__((ext_vector_type(8)))  _Float16 v8h;
typedef __attribute__((ext_vector_type(8)))  float    v8f;

__device__ __forceinline__ v16h frag_a(const _Float16* row, int kk, int half) {
  const v8h lo = *(const v8h*)(row + kk + half * 8);
  const v8h hi = *(const v8h*)(row + kk + 16 + half * 8);
  v16h r;
#pragma unroll
  for (int i = 0; i < 8; ++i) { r[i] = lo[i]; r[i + 8] = hi[i]; }
  return r;
}

__device__ __forceinline__ v16h frag_b(const _Float16* row, int kk, int half) {
  const _Float16* p = row + kk + half * 8;
  const v8h lo = *(const v8h*)(p);
  const v8h hi = *(const v8h*)(p + 16);
  v16h r;
#pragma unroll
  for (int i = 0; i < 8; ++i) { r[i] = lo[i]; r[i + 8] = hi[i]; }
  return r;
}

__device__ __forceinline__ v8f wmma_f16(v16h a, v16h b, v8f c) {
  return __builtin_amdgcn_wmma_f32_16x16x32_f16(false, a, false, b,
                                                (short)0, c, false, false);
}

// Async global->LDS copy of one dword (GLOBAL_LOAD_ASYNC_TO_LDS_B32, GV mode).
__device__ __forceinline__ void async_f32_to_lds(float* lds_dst, const float* src) {
  unsigned off = (unsigned)(unsigned long long)(void*)lds_dst; // LDS byte offset
  asm volatile("global_load_async_to_lds_b32 %0, %1, off"
               :
               : "v"(off), "v"(src)
               : "memory");
}
__device__ __forceinline__ void wait_asynccnt0() {
  asm volatile("s_wait_asynccnt 0x0" ::: "memory");
}

// ---------------------------------------------------------------------------
// Input split: pc[B,N,6] -> xyz f32 [B,N,3], feats f16 [B,N,6] = [pc[3:], xyz]
// ---------------------------------------------------------------------------
__global__ void split_input_kernel(const float* __restrict__ pc,
                                   float* __restrict__ xyz,
                                   _Float16* __restrict__ feats, int total) {
  int t = blockIdx.x * blockDim.x + threadIdx.x;
  if (t >= total) return;
  const float* p = pc + (size_t)t * 6;
  float x = p[0], y = p[1], z = p[2];
  xyz[(size_t)t * 3 + 0] = x;
  xyz[(size_t)t * 3 + 1] = y;
  xyz[(size_t)t * 3 + 2] = z;
  _Float16* f = feats + (size_t)t * 6;
  f[0] = (_Float16)p[3]; f[1] = (_Float16)p[4]; f[2] = (_Float16)p[5];
  f[3] = (_Float16)x;    f[4] = (_Float16)y;    f[5] = (_Float16)z;
}

// ---------------------------------------------------------------------------
// Farthest point sampling: one block per batch, iterative block-wide argmax.
// ---------------------------------------------------------------------------
__global__ __launch_bounds__(256) void fps_kernel(const float* __restrict__ xyz,
                                                  float* __restrict__ dists,
                                                  int* __restrict__ outidx,
                                                  int N, int S) {
  int b = blockIdx.x, tid = threadIdx.x;
  const float* X = xyz + (size_t)b * N * 3;
  float* D = dists + (size_t)b * N;
  __shared__ float rv[256];
  __shared__ int   ri[256];
  __shared__ int   far_s;
  for (int n = tid; n < N; n += 256) D[n] = 1e10f;
  if (tid == 0) far_s = 0;
  __syncthreads();
  for (int i = 0; i < S; ++i) {
    int far = far_s;
    if (tid == 0) outidx[b * S + i] = far;
    float cx = X[far * 3 + 0], cy = X[far * 3 + 1], cz = X[far * 3 + 2];
    float bm = -1.0f; int bi = 0;
    for (int n = tid; n < N; n += 256) {
      float dx = X[n * 3 + 0] - cx;
      float dy = X[n * 3 + 1] - cy;
      float dz = X[n * 3 + 2] - cz;
      float d = dx * dx + dy * dy + dz * dz;
      float old = D[n];
      d = d < old ? d : old;
      D[n] = d;
      if (d > bm) { bm = d; bi = n; }
    }
    rv[tid] = bm; ri[tid] = bi;
    __syncthreads();
    for (int s = 128; s > 0; s >>= 1) {
      if (tid < s && rv[tid + s] > rv[tid]) { rv[tid] = rv[tid + s]; ri[tid] = ri[tid + s]; }
      __syncthreads();
    }
    if (tid == 0) far_s = ri[0];
    __syncthreads();
  }
}

__global__ void gather_centers_kernel(const float* __restrict__ xyz,
                                      const int* __restrict__ idx,
                                      float* __restrict__ nx, int total, int N, int S) {
  int t = blockIdx.x * blockDim.x + threadIdx.x;
  if (t >= total) return;
  int b = t / S;
  int p = idx[t];
  const float* src = xyz + ((size_t)b * N + p) * 3;
  nx[(size_t)t * 3 + 0] = src[0];
  nx[(size_t)t * 3 + 1] = src[1];
  nx[(size_t)t * 3 + 2] = src[2];
}

// ---------------------------------------------------------------------------
// Ball query: ordered index scan == reference's sort-then-take semantics.
// ---------------------------------------------------------------------------
__global__ void ball_query_kernel(const float* __restrict__ xyz,
                                  const float* __restrict__ newxyz,
                                  int* __restrict__ gidx,
                                  int total, int N, int S, int ns, float r2) {
  int t = blockIdx.x * blockDim.x + threadIdx.x;
  if (t >= total) return;
  int b = t / S;
  const float* X = xyz + (size_t)b * N * 3;
  float cx = newxyz[(size_t)t * 3 + 0];
  float cy = newxyz[(size_t)t * 3 + 1];
  float cz = newxyz[(size_t)t * 3 + 2];
  int* out = gidx + (size_t)t * ns;
  int cnt = 0, first = -1;
  for (int n = 0; n < N && cnt < ns; ++n) {
    float dx = X[n * 3 + 0] - cx;
    float dy = X[n * 3 + 1] - cy;
    float dz = X[n * 3 + 2] - cz;
    float d2 = dx * dx + dy * dy + dz * dz;
    if (d2 < r2) {
      if (first < 0) first = n;
      out[cnt++] = n;
    }
  }
  int pad = (first < 0) ? (N - 1) : first;
  for (; cnt < ns; ++cnt) out[cnt] = pad;
}

// ---------------------------------------------------------------------------
// Weight transpose + f16 convert: Wt[n][k] = W[k][n], zero pad K to KP.
// ---------------------------------------------------------------------------
__global__ void convert_wt_kernel(const float* __restrict__ W,
                                  _Float16* __restrict__ Wt,
                                  int Cin, int Cout, int KP) {
  int t = blockIdx.x * blockDim.x + threadIdx.x;
  if (t >= Cout * KP) return;
  int n = t / KP, k = t % KP;
  Wt[t] = (_Float16)(k < Cin ? W[(size_t)k * Cout + n] : 0.0f);
}

// ---------------------------------------------------------------------------
// Fused SA group kernel: gather -> relu(X@W1+b1) -> relu(H1@W2+b2) -> max.
// One block (8 wave32 waves) per group. Each wave owns one N-tile column and
// accumulates all MT row-tiles, so one B fragment feeds MT WMMAs.
// ---------------------------------------------------------------------------
template <int NS, int CF, int K1P, int C1, int C2>
__global__ __launch_bounds__(256) void group_mlp_kernel(
    const float* __restrict__ xyz, const _Float16* __restrict__ feats,
    const int* __restrict__ gidx, const float* __restrict__ newxyz,
    const _Float16* __restrict__ w1t, const float* __restrict__ b1,
    const _Float16* __restrict__ w2t, const float* __restrict__ b2,
    _Float16* __restrict__ fout, int N, int S) {
  constexpr int CIN = 3 + CF;
  constexpr int MT = NS / 16;
  __shared__ __align__(16) _Float16 Xs[NS * K1P];
  __shared__ __align__(16) _Float16 H1s[NS * C1];
  __shared__ unsigned int om[C2];
  __shared__ float b1s[C1];
  __shared__ float b2s[C2];

  int blk = blockIdx.x;
  int b = blk / S;
  int tid = threadIdx.x;
  const int* gi = gidx + (size_t)blk * NS;
  float cx = newxyz[(size_t)blk * 3 + 0];
  float cy = newxyz[(size_t)blk * 3 + 1];
  float cz = newxyz[(size_t)blk * 3 + 2];

  // Stage bias vectors into LDS via async global->LDS DMA (ASYNCcnt path).
  for (int i = tid; i < C1; i += 256) async_f32_to_lds(&b1s[i], &b1[i]);
  for (int i = tid; i < C2; i += 256) async_f32_to_lds(&b2s[i], &b2[i]);
  for (int i = tid; i < C2; i += 256) om[i] = 0u;

  // Gather neighborhood into LDS: X[s][c] = [rel_xyz | feats], K zero-padded.
  for (int i = tid; i < NS * K1P; i += 256) {
    int s = i / K1P, c = i % K1P;
    float v = 0.0f;
    int p = gi[s];
    if (c < 3) {
      float cc = (c == 0) ? cx : ((c == 1) ? cy : cz);
      v = xyz[((size_t)b * N + p) * 3 + c] - cc;
    } else if (c < CIN) {
      v = (float)feats[((size_t)b * N + p) * CF + (c - 3)];
    }
    Xs[i] = (_Float16)v;
  }
  wait_asynccnt0();
  __syncthreads();

  int lane = tid & 31, wave = tid >> 5;
  int half = lane >> 4, l16 = lane & 15;

  // Layer 1: H1 = relu(X @ W1 + b1); wave-per-N-column, MT accumulators.
  {
    constexpr int NT = C1 / 16;
    for (int nt = wave; nt < NT; nt += 8) {
      int n0 = nt * 16;
      float bv = b1s[n0 + l16];
      v8f acc[MT];
#pragma unroll
      for (int mi = 0; mi < MT; ++mi)
#pragma unroll
        for (int i = 0; i < 8; ++i) acc[mi][i] = bv;
      const _Float16* brow = &w1t[(size_t)(n0 + l16) * K1P];
      __builtin_prefetch(brow, 0, 1);
#pragma unroll
      for (int kk = 0; kk < K1P; kk += 32) {
        v16h w = frag_b(brow, kk, half);
#pragma unroll
        for (int mi = 0; mi < MT; ++mi) {
          v16h a = frag_a(&Xs[(mi * 16 + l16) * K1P], kk, half);
          acc[mi] = wmma_f16(a, w, acc[mi]);
        }
      }
#pragma unroll
      for (int mi = 0; mi < MT; ++mi)
#pragma unroll
        for (int r = 0; r < 8; ++r) {
          float v = acc[mi][r];
          v = v > 0.0f ? v : 0.0f;
          H1s[(mi * 16 + r + half * 8) * C1 + n0 + l16] = (_Float16)v;
        }
    }
  }
  __syncthreads();

  // Layer 2: relu(H1 @ W2 + b2) with in-register max over all NS rows.
  {
    constexpr int NT = C2 / 16;
    for (int nt = wave; nt < NT; nt += 8) {
      int n0 = nt * 16;
      float bv = b2s[n0 + l16];
      v8f acc[MT];
#pragma unroll
      for (int mi = 0; mi < MT; ++mi)
#pragma unroll
        for (int i = 0; i < 8; ++i) acc[mi][i] = bv;
      const _Float16* brow = &w2t[(size_t)(n0 + l16) * C1];
      __builtin_prefetch(brow, 0, 1);
#pragma unroll
      for (int kk = 0; kk < C1; kk += 32) {
        v16h w = frag_b(brow, kk, half);
#pragma unroll
        for (int mi = 0; mi < MT; ++mi) {
          v16h a = frag_a(&H1s[(mi * 16 + l16) * C1], kk, half);
          acc[mi] = wmma_f16(a, w, acc[mi]);
        }
      }
      float mx = 0.0f;
#pragma unroll
      for (int mi = 0; mi < MT; ++mi)
#pragma unroll
        for (int r = 0; r < 8; ++r) {
          float v = acc[mi][r];
          v = v > 0.0f ? v : 0.0f;
          mx = v > mx ? v : mx;
        }
      // lanes l and l+16 cover disjoint row halves; post-ReLU >= 0 so uint
      // bit-pattern max == float max
      atomicMax(&om[n0 + l16], __float_as_uint(mx));
    }
  }
  __syncthreads();

  for (int c = tid; c < C2; c += 256)
    fout[(size_t)blk * C2 + c] = (_Float16)__uint_as_float(om[c]);
}

// ---------------------------------------------------------------------------
// Generic WMMA GEMM: out[M,N] = act(A[M,KP] @ Wt[N,KP]^T + bias), f16 in/out.
// Work item = (N-tile, chunk of MCH M-tiles), MCH compile-time and required to
// divide M/16 exactly -> no guards inside the K loop, EXEC untouched.
// ---------------------------------------------------------------------------
template <int MCH, int RELU>
__global__ __launch_bounds__(256) void wmma_gemm_kernel(
    const _Float16* __restrict__ A, const _Float16* __restrict__ Wt,
    const float* __restrict__ bias, _Float16* __restrict__ out,
    int M, int KP, int Nn) {
  int lane = threadIdx.x & 31, wave = threadIdx.x >> 5;
  int half = lane >> 4, l16 = lane & 15;
  int MT = M >> 4, NT = Nn >> 4;
  int MC = MT / MCH;                         // exact by construction
  int work = NT * MC;
  for (int wk = blockIdx.x * 8 + wave; wk < work; wk += gridDim.x * 8) {
    int nt = wk % NT;
    int mb = (wk / NT) * MCH;
    int n0 = nt * 16;
    float bv = bias[n0 + l16];
    v8f acc[MCH];
#pragma unroll
    for (int mi = 0; mi < MCH; ++mi)
#pragma unroll
      for (int i = 0; i < 8; ++i) acc[mi][i] = bv;
    const _Float16* brow = Wt + (size_t)(n0 + l16) * KP;
    __builtin_prefetch(brow, 0, 1);
    for (int kk = 0; kk < KP; kk += 32) {
      v16h w = frag_b(brow, kk, half);
#pragma unroll
      for (int mi = 0; mi < MCH; ++mi) {
        v16h a = frag_a(A + (size_t)((mb + mi) * 16 + l16) * KP, kk, half);
        acc[mi] = wmma_f16(a, w, acc[mi]);
      }
    }
#pragma unroll
    for (int mi = 0; mi < MCH; ++mi) {
#pragma unroll
      for (int r = 0; r < 8; ++r) {
        float v = acc[mi][r];
        if (RELU) v = v > 0.0f ? v : 0.0f;
        out[(size_t)((mb + mi) * 16 + r + half * 8) * Nn + n0 + l16] = (_Float16)v;
      }
    }
  }
}

// ---------------------------------------------------------------------------
// Head helpers
// ---------------------------------------------------------------------------
__global__ void assemble_final_kernel(const float* __restrict__ nx3,
                                      const _Float16* __restrict__ feats3,
                                      _Float16* __restrict__ A0, int total) {
  int t = blockIdx.x * blockDim.x + threadIdx.x;
  if (t >= total) return;
  int row = t / 544, c = t % 544;
  float v = 0.0f;
  if (c < 3) v = nx3[(size_t)row * 3 + c];
  else if (c < 515) v = (float)feats3[(size_t)row * 512 + (c - 3)];
  A0[t] = (_Float16)v;
}

__global__ void maxpts_kernel(const _Float16* __restrict__ G,
                              _Float16* __restrict__ g, int total, int P, int C) {
  int t = blockIdx.x * blockDim.x + threadIdx.x;
  if (t >= total) return;
  int b = t / C, c = t % C;
  float m = -1e30f;
  for (int s = 0; s < P; ++s) {
    float v = (float)G[((size_t)b * P + s) * C + c];
    m = v > m ? v : m;
  }
  g[t] = (_Float16)m;
}

__global__ void finalize_kernel(const _Float16* __restrict__ F2,
                                const float* __restrict__ gamma,
                                const float* __restrict__ beta,
                                float* __restrict__ out, int total, int C) {
  int t = blockIdx.x * blockDim.x + threadIdx.x;
  if (t >= total) return;
  int c = t % C;
  float inv = rsqrtf(1.0f + 1e-5f);
  out[t] = (float)F2[t] * inv * gamma[c] + beta[c];
}

// ---------------------------------------------------------------------------
// Host launch
// ---------------------------------------------------------------------------
extern "C" void kernel_launch(void* const* d_in, const int* in_sizes, int n_in,
                              void* d_out, int out_size, void* d_ws, size_t ws_size,
                              hipStream_t stream) {
  (void)in_sizes; (void)n_in; (void)out_size; (void)ws_size;
  const int B = 32, N1 = 4096;
  const int S1 = 256, S2 = 64, S3 = 16;
  const int NS1 = 64, NS2 = 64, NS3 = 32;

  const float* pc    = (const float*)d_in[0];
  const float* w1a   = (const float*)d_in[1];
  const float* b1a   = (const float*)d_in[2];
  const float* w1b   = (const float*)d_in[3];
  const float* b1b   = (const float*)d_in[4];
  const float* w2a   = (const float*)d_in[5];
  const float* b2a   = (const float*)d_in[6];
  const float* w2b   = (const float*)d_in[7];
  const float* b2b   = (const float*)d_in[8];
  const float* w3a   = (const float*)d_in[9];
  const float* b3a   = (const float*)d_in[10];
  const float* w3b   = (const float*)d_in[11];
  const float* b3b   = (const float*)d_in[12];
  const float* w4a   = (const float*)d_in[13];
  const float* b4a   = (const float*)d_in[14];
  const float* w4b   = (const float*)d_in[15];
  const float* b4b   = (const float*)d_in[16];
  const float* fc1w  = (const float*)d_in[17];
  const float* fc1b  = (const float*)d_in[18];
  const float* fc2w  = (const float*)d_in[19];
  const float* fc2b  = (const float*)d_in[20];
  const float* gamma = (const float*)d_in[21];
  const float* beta  = (const float*)d_in[22];
  float* yout = (float*)d_out;

  char* ws = (char*)d_ws;
  size_t off = 0;
  auto alloc = [&](size_t bytes) -> void* {
    void* p = ws + off;
    off += (bytes + 255) & ~(size_t)255;
    return p;
  };

  float*     xyz0   = (float*)alloc((size_t)B * N1 * 3 * 4);
  _Float16*  feats0 = (_Float16*)alloc((size_t)B * N1 * 6 * 2);
  float*     dists  = (float*)alloc((size_t)B * N1 * 4);
  int*       fps1   = (int*)alloc((size_t)B * S1 * 4);
  float*     nx1    = (float*)alloc((size_t)B * S1 * 3 * 4);
  int*       gidx1  = (int*)alloc((size_t)B * S1 * NS1 * 4);
  _Float16*  feats1 = (_Float16*)alloc((size_t)B * S1 * 128 * 2);
  int*       fps2   = (int*)alloc((size_t)B * S2 * 4);
  float*     nx2    = (float*)alloc((size_t)B * S2 * 3 * 4);
  int*       gidx2  = (int*)alloc((size_t)B * S2 * NS2 * 4);
  _Float16*  feats2 = (_Float16*)alloc((size_t)B * S2 * 256 * 2);
  int*       fps3   = (int*)alloc((size_t)B * S3 * 4);
  float*     nx3    = (float*)alloc((size_t)B * S3 * 3 * 4);
  int*       gidx3  = (int*)alloc((size_t)B * S3 * NS3 * 4);
  _Float16*  feats3 = (_Float16*)alloc((size_t)B * S3 * 512 * 2);
  _Float16*  A0     = (_Float16*)alloc((size_t)512 * 544 * 2);
  _Float16*  G1     = (_Float16*)alloc((size_t)512 * 512 * 2);
  _Float16*  G2     = (_Float16*)alloc((size_t)512 * 1024 * 2);
  _Float16*  gfeat  = (_Float16*)alloc((size_t)32 * 1024 * 2);
  _Float16*  F1     = (_Float16*)alloc((size_t)32 * 1024 * 2);
  _Float16*  F2     = (_Float16*)alloc((size_t)32 * 128 * 2);
  _Float16*  w1at   = (_Float16*)alloc((size_t)64 * 32 * 2);
  _Float16*  w1bt   = (_Float16*)alloc((size_t)128 * 64 * 2);
  _Float16*  w2at   = (_Float16*)alloc((size_t)128 * 160 * 2);
  _Float16*  w2bt   = (_Float16*)alloc((size_t)256 * 128 * 2);
  _Float16*  w3at   = (_Float16*)alloc((size_t)256 * 288 * 2);
  _Float16*  w3bt   = (_Float16*)alloc((size_t)512 * 256 * 2);
  _Float16*  w4at   = (_Float16*)alloc((size_t)512 * 544 * 2);
  _Float16*  w4bt   = (_Float16*)alloc((size_t)1024 * 512 * 2);
  _Float16*  fc1t   = (_Float16*)alloc((size_t)1024 * 1024 * 2);
  _Float16*  fc2t   = (_Float16*)alloc((size_t)128 * 1024 * 2);

  auto cvt = [&](const float* W, _Float16* Wt, int Cin, int Cout, int KP) {
    int tot = Cout * KP;
    convert_wt_kernel<<<(tot + 255) / 256, 256, 0, stream>>>(W, Wt, Cin, Cout, KP);
  };
  cvt(w1a, w1at, 9, 64, 32);
  cvt(w1b, w1bt, 64, 128, 64);
  cvt(w2a, w2at, 131, 128, 160);
  cvt(w2b, w2bt, 128, 256, 128);
  cvt(w3a, w3at, 259, 256, 288);
  cvt(w3b, w3bt, 256, 512, 256);
  cvt(w4a, w4at, 515, 512, 544);
  cvt(w4b, w4bt, 512, 1024, 512);
  cvt(fc1w, fc1t, 1024, 1024, 1024);
  cvt(fc2w, fc2t, 1024, 128, 1024);

  {
    int tot = B * N1;
    split_input_kernel<<<(tot + 255) / 256, 256, 0, stream>>>(pc, xyz0, feats0, tot);
  }

  // ---- SA1: 4096 -> 256 centers, r=0.1, ns=64, MLP 9->64->128 ----
  fps_kernel<<<B, 256, 0, stream>>>(xyz0, dists, fps1, N1, S1);
  gather_centers_kernel<<<(B * S1 + 255) / 256, 256, 0, stream>>>(xyz0, fps1, nx1, B * S1, N1, S1);
  ball_query_kernel<<<(B * S1 + 255) / 256, 256, 0, stream>>>(xyz0, nx1, gidx1, B * S1, N1, S1, NS1, 0.01f);
  group_mlp_kernel<64, 6, 32, 64, 128><<<B * S1, 256, 0, stream>>>(
      xyz0, feats0, gidx1, nx1, w1at, b1a, w1bt, b1b, feats1, N1, S1);

  // ---- SA2: 256 -> 64 centers, r=0.2, ns=64, MLP 131->128->256 ----
  fps_kernel<<<B, 256, 0, stream>>>(nx1, dists, fps2, S1, S2);
  gather_centers_kernel<<<(B * S2 + 255) / 256, 256, 0, stream>>>(nx1, fps2, nx2, B * S2, S1, S2);
  ball_query_kernel<<<(B * S2 + 255) / 256, 256, 0, stream>>>(nx1, nx2, gidx2, B * S2, S1, S2, NS2, 0.04f);
  group_mlp_kernel<64, 128, 160, 128, 256><<<B * S2, 256, 0, stream>>>(
      nx1, feats1, gidx2, nx2, w2at, b2a, w2bt, b2b, feats2, S1, S2);

  // ---- SA3: 64 -> 16 centers, r=0.4, ns=32, MLP 259->256->512 ----
  fps_kernel<<<B, 256, 0, stream>>>(nx2, dists, fps3, S2, S3);
  gather_centers_kernel<<<(B * S3 + 255) / 256, 256, 0, stream>>>(nx2, fps3, nx3, B * S3, S2, S3);
  ball_query_kernel<<<(B * S3 + 255) / 256, 256, 0, stream>>>(nx2, nx3, gidx3, B * S3, S2, S3, NS3, 0.16f);
  group_mlp_kernel<32, 256, 288, 256, 512><<<B * S3, 256, 0, stream>>>(
      nx2, feats2, gidx3, nx3, w3at, b3a, w3bt, b3b, feats3, S2, S3);

  // ---- Head: concat -> 515->512->1024 relu MLP -> max over 16 pts -> FCs ----
  {
    int tot = 512 * 544;
    assemble_final_kernel<<<(tot + 255) / 256, 256, 0, stream>>>(nx3, feats3, A0, tot);
  }
  {
    int work = (512 / 16) * ((512 / 16) / 4);
    wmma_gemm_kernel<4, 1><<<(work + 7) / 8, 256, 0, stream>>>(A0, w4at, b4a, G1, 512, 544, 512);
  }
  {
    int work = (1024 / 16) * ((512 / 16) / 4);
    wmma_gemm_kernel<4, 1><<<(work + 7) / 8, 256, 0, stream>>>(G1, w4bt, b4b, G2, 512, 512, 1024);
  }
  {
    int tot = 32 * 1024;
    maxpts_kernel<<<(tot + 255) / 256, 256, 0, stream>>>(G2, gfeat, tot, 16, 1024);
  }
  {
    int work = (1024 / 16) * ((32 / 16) / 2);
    wmma_gemm_kernel<2, 1><<<(work + 7) / 8, 256, 0, stream>>>(gfeat, fc1t, fc1b, F1, 32, 1024, 1024);
  }
  {
    int work = (128 / 16) * ((32 / 16) / 2);
    wmma_gemm_kernel<2, 0><<<(work + 7) / 8, 256, 0, stream>>>(F1, fc2t, fc2b, F2, 32, 1024, 128);
  }
  {
    int tot = 32 * 128;
    finalize_kernel<<<(tot + 255) / 256, 256, 0, stream>>>(F2, gamma, beta, yout, tot, 128);
  }
}
